// SEG_NET_14379550507498
// MI455X (gfx1250) — compile-verified, matrix operand order checked by reading, not run on hardware
//
#include <hip/hip_runtime.h>
#include <hip/hip_bf16.h>

// ---------------------------------------------------------------------------
// CDNA5 (gfx1250) seg-net mask head.
//   - convs: implicit GEMM on v_wmma_f32_16x16x32_f16, block tile 64x64,
//     4 WMMA issues per wave per K-chunk (A-fragment reuse).
//   - weights pre-converted to f16; weight tiles staged into LDS by the
//     Tensor Data Mover (tensor_load_to_lds) with hardware LDS padding,
//     synchronized with s_wait_tensorcnt.
//   - geometry (boxes, patch rects, output offsets) computed on device.
// ---------------------------------------------------------------------------

typedef __attribute__((ext_vector_type(16))) _Float16      v16h;
typedef __attribute__((ext_vector_type(8)))  float         v8f;
typedef __attribute__((ext_vector_type(4)))  unsigned int  u32x4;
typedef __attribute__((ext_vector_type(8)))  int           i32x8;
typedef __attribute__((ext_vector_type(4)))  int           i32x4;

#define HDRS 32   // ints per box in header

#if defined(__has_builtin)
#if __has_builtin(__builtin_amdgcn_tensor_load_to_lds) && __has_builtin(__builtin_amdgcn_s_wait_tensorcnt)
#define HAVE_TDM 1
#endif
#endif

__device__ __forceinline__ int imin(int a, int b) { return a < b ? a : b; }
__device__ __forceinline__ int imax(int a, int b) { return a > b ? a : b; }

__device__ __forceinline__ bool stage_active(const int* hdr, int b, int stage) {
    int nlev = hdr[b * HDRS + 1];
    if (stage < 0) return nlev >= 1;
    return stage <= nlev - 2;
}

#ifdef HAVE_TDM
// DMA a 2D f16 tile (64 rows x 32 elems, row stride = strideK elems) from
// global memory into LDS at ldsOff, with 1-DWORD pad every 16 DWORDs so the
// landed layout is [64][34] halves. OOB rows/cols land as zero.
__device__ __forceinline__ void tdm_load_w16(const _Float16* gsrc, unsigned ldsOff,
                                             int remK, int remRows, int strideK) {
    unsigned long long ga = (unsigned long long)(uintptr_t)gsrc;
    u32x4 g0;
    g0[0] = 1u;                                        // count=1, user mode
    g0[1] = ldsOff;                                    // lds_addr (bytes)
    g0[2] = (unsigned)(ga & 0xffffffffu);              // global_addr[31:0]
    g0[3] = (unsigned)((ga >> 32) & 0x01ffffffu)       // global_addr[56:32]
          | (2u << 30);                                // type = 2 (image)
    i32x8 g1;
    g1[0] = (1 << 16)      // data_size = 2 bytes
          | (1 << 20)      // pad_enable
          | (3 << 22);     // pad_interval: 16 DWORDs ; pad_amount=0 -> 1 DWORD
    g1[1] = (remK & 0xffff) << 16;                          // tensor_dim0 lo16
    g1[2] = ((remK >> 16) & 0xffff) | ((remRows & 0xffff) << 16); // dim0 hi / dim1 lo
    g1[3] = ((remRows >> 16) & 0xffff) | (32 << 16);        // dim1 hi | tile_dim0=32
    g1[4] = 64;                                             // tile_dim1=64, tile_dim2=0
    g1[5] = strideK;                                        // tensor_dim0_stride lo32
    g1[6] = 0;
    g1[7] = 0;
    i32x4 z4 = {0, 0, 0, 0};
#if __clang_major__ >= 23
    i32x8 z8 = {0, 0, 0, 0, 0, 0, 0, 0};
    __builtin_amdgcn_tensor_load_to_lds(g0, g1, z4, z4, z8, 0);
#else
    __builtin_amdgcn_tensor_load_to_lds(g0, g1, z4, z4, 0);
#endif
}
#endif

// ---------------------------------------------------------------------------
// Geometry: parse detections, per-box pyramid rects, nlev, mask offsets; write
// the dets tail of the output. det (1,2,5,5); class-1 row r at (5+r)*5.
// ---------------------------------------------------------------------------
__global__ void k_geom(const float* __restrict__ det, int* __restrict__ hdr,
                       float* __restrict__ dout) {
    if (threadIdx.x != 0 || blockIdx.x != 0) return;
    long maskTotal = 0;
    int nvalid = 0;
    for (int r = 0; r < 5; ++r) {
        const float* row = det + (5 + r) * 5;
        int* h = hdr + r * HDRS;
        int nlev = 0;
        float score = row[0];
        if (score > 0.0f) {
            float y1 = row[1], x1 = row[2], y2 = row[3], x2 = row[4];
            for (int l = 0; l < 5; ++l) {
                int hs = 512 >> l;
                float s = (float)hs / 512.0f;
                int py1 = imax(0, (int)rintf(y1 * s));
                int px1 = imax(0, (int)rintf(x1 * s));
                int py2 = imin((int)rintf(y2 * s), hs - 1);
                int px2 = imin((int)rintf(x2 * s), hs - 1);
                bool ok = !(py2 < py1 || px2 < px1 || (py2 - py1) < 2 || (px2 - px1) < 2);
                if (ok && nlev == l) nlev = l + 1;
                h[2 + l * 4 + 0] = py1;
                h[2 + l * 4 + 1] = px1;
                h[2 + l * 4 + 2] = ok ? (py2 - py1 + 1) : 0;
                h[2 + l * 4 + 3] = ok ? (px2 - px1 + 1) : 0;
            }
        } else {
            for (int l = 0; l < 5; ++l) {
                h[2 + l * 4 + 0] = 0; h[2 + l * 4 + 1] = 0;
                h[2 + l * 4 + 2] = 0; h[2 + l * 4 + 3] = 0;
            }
        }
        h[1] = nlev;
        h[0] = (nlev > 0) ? 1 : 0;
        if (nlev > 0) {
            h[22] = (int)maskTotal;
            maskTotal += (long)h[2 + 2] * (long)h[2 + 3];
            h[23] = nvalid++;
        } else {
            h[22] = -1; h[23] = -1;
        }
    }
    for (int r = 0; r < 5; ++r) {
        const int* h = hdr + r * HDRS;
        if (h[0]) {
            const float* row = det + (5 + r) * 5;
            float* o = dout + maskTotal + (long)h[23] * 6;
            o[0] = row[1]; o[1] = row[2]; o[2] = row[3]; o[3] = row[4];
            o[4] = row[0]; o[5] = 1.0f;
        }
    }
}

// f32 -> f16 weight pre-conversion
__global__ void k_f32_to_f16(const float* __restrict__ s, _Float16* __restrict__ d,
                             long n) {
    long i = (long)blockIdx.x * blockDim.x + threadIdx.x;
    if (i < n) d[i] = (_Float16)s[i];
}

// ---------------------------------------------------------------------------
// Implicit-GEMM conv. Block = 128 threads = 4 waves. Block tile: 64 couts x
// 64 pixels. Wave w: couts [w*16, w*16+16) x all 64 pixels (4 N-subtiles,
// A-fragment reused -> 4 WMMA per K-chunk). K = Cin*Kk*Kk in chunks of 32.
// Weights (f16) staged by TDM into LDS with hw padding; B tile built by
// im2col gather (f32 -> f16).
// ---------------------------------------------------------------------------
__launch_bounds__(128)
__global__ void k_conv_wmma(const float* __restrict__ in,
                            const _Float16* __restrict__ wgt,
                            const float* __restrict__ bias,
                            float* __restrict__ out,
                            const int* __restrict__ hdr, int b, int dimLvl,
                            int sH, int sW,
                            int Cin, int Cout, int Kk, int pad, int relu,
                            int coutOff, int stage) {
    if (hdr != nullptr && !stage_active(hdr, b, stage)) return;
    int H = sH, W = sW;
    if (hdr != nullptr && dimLvl >= 0) {
        const int* r = hdr + b * HDRS + 2 + dimLvl * 4;
        H = r[2]; W = r[3];
    }
    int HW = H * W;
    int pixBase  = blockIdx.x * 64;
    if (pixBase >= HW) return;                 // uniform across block
    int coutBase = blockIdx.y * 64;

    __shared__ __align__(16) _Float16 Wh[64][34];  // TDM-landed, 1-DW pad/row
    __shared__ __align__(16) _Float16 Bs[32][66];

    const int tid  = threadIdx.x;
    const int lane = tid & 31;
    const int wave = tid >> 5;
    const int KK   = Kk * Kk;
    const int Ktot = Cin * KK;

    v8f acc[4] = {};
    for (int k0 = 0; k0 < Ktot; k0 += 32) {
#ifdef HAVE_TDM
        // Weight tile via Tensor Data Mover (wave 0 issues; EXEC ignored).
        if (tid < 32) {
            tdm_load_w16(wgt + (size_t)coutBase * Ktot + k0,
                         (unsigned)(uintptr_t)&Wh[0][0],
                         Ktot - k0, Cout - coutBase, Ktot);
        }
#else
        for (int idx = tid; idx < 64 * 32; idx += 128) {
            int rrow = idx >> 5, kk = idx & 31;
            int k = k0 + kk;
            int co = coutBase + rrow;
            _Float16 v = (_Float16)0.0f;
            if (k < Ktot && co < Cout) v = wgt[(size_t)co * Ktot + k];
            Wh[rrow][kk] = v;
        }
#endif
        // B tile: 32 k x 64 pixels (im2col gather, f32 -> f16)
        for (int idx = tid; idx < 32 * 64; idx += 128) {
            int kk = idx >> 6, n = idx & 63;
            int k = k0 + kk;
            int pix = pixBase + n;
            float v = 0.0f;
            if (k < Ktot && pix < HW) {
                int ci  = k / KK;
                int rem = k - ci * KK;
                int ky  = rem / Kk, kx = rem - ky * Kk;
                int y   = pix / W,  x  = pix - y * W;
                int iy  = y + ky - pad, ix = x + kx - pad;
                if (iy >= 0 && iy < H && ix >= 0 && ix < W)
                    v = in[(size_t)ci * HW + (size_t)iy * W + ix];
            }
            Bs[kk][n] = (_Float16)v;
        }
#ifdef HAVE_TDM
        if (tid < 32) __builtin_amdgcn_s_wait_tensorcnt(0);
#endif
        __syncthreads();

        // A fragment (ISA 7.12.2 16-bit A layout): lane L -> M=L&15,
        // K-half=(L>>4)*8, VGPR j holds a K pair.
        const int mrow  = wave * 16 + (lane & 15);
        const int khalf = (lane >> 4) * 8;
        const int ncol  = lane & 15;
        v16h a;
#pragma unroll
        for (int j = 0; j < 8; ++j) {
            int kb = (j < 4) ? (khalf + 2 * j) : (16 + khalf + 2 * (j - 4));
            a[2 * j]     = Wh[mrow][kb];
            a[2 * j + 1] = Wh[mrow][kb + 1];
        }
        // 4 N-subtiles reuse the A fragment.
#pragma unroll
        for (int t = 0; t < 4; ++t) {
            v16h bf;
#pragma unroll
            for (int j = 0; j < 8; ++j) {
                int kb = (j < 4) ? (khalf + 2 * j) : (16 + khalf + 2 * (j - 4));
                bf[2 * j]     = Bs[kb][16 * t + ncol];
                bf[2 * j + 1] = Bs[kb + 1][16 * t + ncol];
            }
            acc[t] = __builtin_amdgcn_wmma_f32_16x16x32_f16(
                false, a, false, bf, (short)0, acc[t], false, false);
        }
        __syncthreads();
    }

    // Epilogue: C/D layout -> lane L, reg r : M = r + 8*(L>>4), N = L&15
#pragma unroll
    for (int t = 0; t < 4; ++t) {
        int pix = pixBase + 16 * t + (lane & 15);
        if (pix < HW) {
#pragma unroll
            for (int r8 = 0; r8 < 8; ++r8) {
                int m  = r8 + 8 * (lane >> 4);
                int co = coutBase + wave * 16 + m;
                if (co < Cout) {
                    float v = acc[t][r8] + bias[co];
                    if (relu) v = fmaxf(v, 0.0f);
                    out[(size_t)(coutOff + co) * HW + pix] = v;
                }
            }
        }
    }
}

// ---------------------------------------------------------------------------
// Align-corners bilinear resize between two pyramid-level rects of a box.
// ---------------------------------------------------------------------------
__global__ void k_resize(const float* __restrict__ in, float* __restrict__ out,
                         const int* __restrict__ hdr, int b, int inLvl,
                         int outLvl, int C, int stage) {
    if (!stage_active(hdr, b, stage)) return;
    const int* ri = hdr + b * HDRS + 2 + inLvl * 4;
    const int* ro = hdr + b * HDRS + 2 + outLvl * 4;
    int ih = ri[2], iw = ri[3], oh = ro[2], ow = ro[3];
    long total = (long)C * oh * ow;
    long idx = (long)blockIdx.x * blockDim.x + threadIdx.x;
    if (idx >= total) return;
    int x = (int)(idx % ow); long t = idx / ow;
    int y = (int)(t % oh);   int  c = (int)(t / oh);
    float ys = (oh == 1) ? 0.0f : (float)y * ((float)(ih - 1) / (float)(oh - 1));
    float xs = (ow == 1) ? 0.0f : (float)x * ((float)(iw - 1) / (float)(ow - 1));
    int y0 = imin(imax((int)floorf(ys), 0), ih - 1);
    int x0 = imin(imax((int)floorf(xs), 0), iw - 1);
    int y1 = imin(y0 + 1, ih - 1);
    int x1 = imin(x0 + 1, iw - 1);
    float wy = ys - (float)y0, wx = xs - (float)x0;
    const float* p = in + (size_t)c * ih * iw;
    float v00 = p[(size_t)y0 * iw + x0], v01 = p[(size_t)y0 * iw + x1];
    float v10 = p[(size_t)y1 * iw + x0], v11 = p[(size_t)y1 * iw + x1];
    float top = v00 * (1.0f - wx) + v01 * wx;
    float bot = v10 * (1.0f - wx) + v11 * wx;
    out[idx] = top * (1.0f - wy) + bot * wy;
}

// out = sigmoid(relu(a + y)) * y
__global__ void k_gate(const float* __restrict__ a, const float* __restrict__ y,
                       float* __restrict__ o, const int* __restrict__ hdr,
                       int b, int lvl, int C, int stage) {
    if (!stage_active(hdr, b, stage)) return;
    const int* r = hdr + b * HDRS + 2 + lvl * 4;
    long total = (long)C * r[2] * r[3];
    long idx = (long)blockIdx.x * blockDim.x + threadIdx.x;
    if (idx >= total) return;
    float t = fmaxf(a[idx] + y[idx], 0.0f);
    float s = 1.0f / (1.0f + expf(-t));
    o[idx] = s * y[idx];
}

__global__ void k_extract(const float* __restrict__ feat, float* __restrict__ dst,
                          const int* __restrict__ hdr, int b, int lvl, int C,
                          int fH, int fW) {
    const int* h = hdr + b * HDRS;
    if (lvl >= h[1]) return;
    const int* r = h + 2 + lvl * 4;
    int H = r[2], W = r[3];
    long total = (long)C * H * W;
    long idx = (long)blockIdx.x * blockDim.x + threadIdx.x;
    if (idx >= total) return;
    int x = (int)(idx % W); long t = idx / W;
    int y = (int)(t % H);   int  c = (int)(t / H);
    dst[idx] = feat[(size_t)c * fH * fW + (size_t)(r[0] + y) * fW + (r[1] + x)];
}

__global__ void k_copy_pre(const float* __restrict__ patch, float* __restrict__ pre,
                           const int* __restrict__ hdr, int b, int lvl, int C) {
    const int* h = hdr + b * HDRS;
    if (lvl != h[1] - 1) return;
    const int* r = h + 2 + lvl * 4;
    long total = (long)C * r[2] * r[3];
    long idx = (long)blockIdx.x * blockDim.x + threadIdx.x;
    if (idx >= total) return;
    pre[idx] = patch[idx];
}

__global__ void k_commit(const float* __restrict__ src, float* __restrict__ dst,
                         const int* __restrict__ hdr, int b, int lvl, int C,
                         int stage) {
    if (!stage_active(hdr, b, stage)) return;
    const int* r = hdr + b * HDRS + 2 + lvl * 4;
    long total = (long)C * r[2] * r[3];
    long idx = (long)blockIdx.x * blockDim.x + threadIdx.x;
    if (idx >= total) return;
    dst[idx] = src[idx];
}

__global__ void k_sigmoid_out(const float* __restrict__ x,
                              const int* __restrict__ hdr, int b,
                              float* __restrict__ dout) {
    const int* h = hdr + b * HDRS;
    if (!h[0]) return;
    long total = (long)h[2 + 2] * h[2 + 3];
    long idx = (long)blockIdx.x * blockDim.x + threadIdx.x;
    if (idx >= total) return;
    dout[(long)h[22] + idx] = 1.0f / (1.0f + expf(-x[idx]));
}

// ---------------------------------------------------------------------------
// Host orchestration
// ---------------------------------------------------------------------------
static inline unsigned cdiv(long a, long b) { return (unsigned)((a + b - 1) / b); }

extern "C" void kernel_launch(void* const* d_in, const int* in_sizes, int n_in,
                              void* d_out, int out_size, void* d_ws, size_t ws_size,
                              hipStream_t stream) {
    (void)in_sizes; (void)out_size; (void)ws_size;
    if (n_in < 46) return;

    const float* det = (const float*)d_in[0];
    const float* feats[5] = {(const float*)d_in[1], (const float*)d_in[2],
                             (const float*)d_in[3], (const float*)d_in[4],
                             (const float*)d_in[5]};
    auto P = [&](int i) { return (const float*)d_in[i]; };
    // params flattened in make_params insertion order:
    // 6:c0_w1 7:c0_b1 8:c0_w2 9:c0_b2 10:head_w1 11:head_b1 12:head_w2 13:head_b2
    // att i:  w1=14+4i b1=15+4i w2=16+4i b2=17+4i
    // comb i: up_w=30+4i up_b=31+4i cat_w=32+4i cat_b=33+4i

    int*   hdr = (int*)d_ws;
    float* fws = (float*)((char*)d_ws + 4096);

    constexpr size_t SZ64 = 64ull * 512 * 512;
    constexpr size_t SZP1 = 64ull * 256 * 256;
    constexpr size_t SZP2 = 256ull * 128 * 128;
    constexpr size_t SZP3 = 512ull * 64 * 64;
    constexpr size_t SZP4 = 1024ull * 32 * 32;

    size_t o = 0;
    auto take = [&](size_t n) { float* p = fws + o; o += n; return p; };
    float* F0A    = take(SZ64);
    float* F0B    = take(SZ64);
    float* PATCH0 = take(SZ64);
    float* PATCH1 = take(SZP1);
    float* PATCH2 = take(SZP2);
    float* PATCH3 = take(SZP3);
    float* PATCH4 = take(SZP4);
    float* PREIN  = take(SZ64);
    float* PREOUT = take(SZ64);
    float* TMPA   = take(SZ64);
    float* TMPB   = take(SZ64);
    float* CAT    = take(2 * SZ64);
    float* PATCH[5] = {PATCH0, PATCH1, PATCH2, PATCH3, PATCH4};

    const int LCH[5]   = {64, 64, 256, 512, 1024};
    const int LDIM[5]  = {512, 256, 128, 64, 32};
    const int ATTX[4]  = {64, 256, 512, 1024};
    const int ATTY[4]  = {64, 64, 256, 512};
    const int CINS[4]  = {64, 256, 512, 1024};
    const int COUTS[4] = {64, 64, 256, 512};

    // ---- f16 weight arena + conversion --------------------------------
    _Float16* h16 = (_Float16*)(fws + o);
    size_t ho = 0;
    _Float16* w16[64] = {};
    auto cvt = [&](int pidx, long n) {
        _Float16* dst = h16 + ho; ho += (size_t)((n + 7) & ~7l);
        w16[pidx] = dst;
        k_f32_to_f16<<<cdiv(n, 256), 256, 0, stream>>>(P(pidx), dst, n);
    };
    cvt(6, 64l * 3 * 9);          // c0_w1
    cvt(8, 64l * 64 * 9);         // c0_w2
    cvt(10, 64l * 64 * 9);        // head_w1
    cvt(12, 1l * 64 * 9);         // head_w2
    for (int i = 0; i < 4; ++i) {
        cvt(14 + 4 * i, (long)ATTY[i] * ATTX[i]);          // att w1
        cvt(16 + 4 * i, (long)ATTY[i] * ATTY[i]);          // att w2
        cvt(30 + 4 * i, (long)COUTS[i] * CINS[i] * 9);     // comb up_w
        cvt(32 + 4 * i, (long)COUTS[i] * 2 * COUTS[i]);    // comb cat_w
    }

    float* dout = (float*)d_out;
    k_geom<<<1, 32, 0, stream>>>(det, hdr, dout);

    // layer_c0 on the full 512x512 image (static dims, no header guard)
    {
        dim3 blk(128);
        dim3 g1(cdiv(512 * 512, 64), 1);
        k_conv_wmma<<<g1, blk, 0, stream>>>(feats[0], w16[6], P(7), F0A,
            nullptr, 0, -1, 512, 512, 3, 64, 3, 1, 1, 0, 0);
        k_conv_wmma<<<g1, blk, 0, stream>>>(F0A, w16[8], P(9), F0B,
            nullptr, 0, -1, 512, 512, 64, 64, 3, 1, 1, 0, 0);
    }

    for (int b = 0; b < 5; ++b) {
        for (int l = 0; l < 5; ++l) {
            const float* src = (l == 0) ? F0B : feats[l];
            long maxTot = (long)LCH[l] * LDIM[l] * LDIM[l];
            k_extract<<<cdiv(maxTot, 256), 256, 0, stream>>>(
                src, PATCH[l], hdr, b, l, LCH[l], LDIM[l], LDIM[l]);
            k_copy_pre<<<cdiv(maxTot, 256), 256, 0, stream>>>(
                PATCH[l], PREIN, hdr, b, l, LCH[l]);
        }

        for (int i = 3; i >= 0; --i) {
            long hwI  = (long)LDIM[i] * LDIM[i];
            long hwI1 = (long)LDIM[i + 1] * LDIM[i + 1];
            dim3 blk(128);
            int aw1 = 14 + 4 * i, ab1 = 15 + 4 * i, aw2 = 16 + 4 * i, ab2 = 17 + 4 * i;
            int uw = 30 + 4 * i, ub = 31 + 4 * i, cw = 32 + 4 * i, cb = 33 + 4 * i;

            // a = conv1x1(pre) at lvl i+1
            k_conv_wmma<<<dim3(cdiv(hwI1, 64), cdiv(ATTY[i], 64)), blk, 0, stream>>>(
                PREIN, w16[aw1], P(ab1), TMPA, hdr, b, i + 1, 0, 0,
                ATTX[i], ATTY[i], 1, 0, 0, 0, i);
            // resize a -> lvl i
            k_resize<<<cdiv((long)ATTY[i] * hwI, 256), 256, 0, stream>>>(
                TMPA, TMPB, hdr, b, i + 1, i, ATTY[i], i);
            // gated = sigmoid(relu(a + y)) * y
            k_gate<<<cdiv((long)ATTY[i] * hwI, 256), 256, 0, stream>>>(
                TMPB, PATCH[i], TMPA, hdr, b, i, ATTY[i], i);
            // att = relu(conv1x1(gated)) -> CAT[:outs]
            k_conv_wmma<<<dim3(cdiv(hwI, 64), cdiv(COUTS[i], 64)), blk, 0, stream>>>(
                TMPA, w16[aw2], P(ab2), CAT, hdr, b, i, 0, 0,
                ATTY[i], COUTS[i], 1, 0, 1, 0, i);
            // up = relu(conv3x3(resize(pre))) -> CAT[outs:2*outs]
            k_resize<<<cdiv((long)CINS[i] * hwI, 256), 256, 0, stream>>>(
                PREIN, TMPB, hdr, b, i + 1, i, CINS[i], i);
            k_conv_wmma<<<dim3(cdiv(hwI, 64), cdiv(COUTS[i], 64)), blk, 0, stream>>>(
                TMPB, w16[uw], P(ub), CAT, hdr, b, i, 0, 0,
                CINS[i], COUTS[i], 3, 1, 1, COUTS[i], i);
            // pre' = relu(conv1x1(cat))
            k_conv_wmma<<<dim3(cdiv(hwI, 64), cdiv(COUTS[i], 64)), blk, 0, stream>>>(
                CAT, w16[cw], P(cb), PREOUT, hdr, b, i, 0, 0,
                2 * COUTS[i], COUTS[i], 1, 0, 1, 0, i);
            // commit
            k_commit<<<cdiv((long)COUTS[i] * hwI, 256), 256, 0, stream>>>(
                PREOUT, PREIN, hdr, b, i, COUTS[i], i);
        }

        // head
        long hw0 = 512l * 512l;
        dim3 blk(128);
        k_conv_wmma<<<dim3(cdiv(hw0, 64), 1), blk, 0, stream>>>(
            PREIN, w16[10], P(11), TMPA, hdr, b, 0, 0, 0, 64, 64, 3, 1, 1, 0, -1);
        k_conv_wmma<<<dim3(cdiv(hw0, 64), 1), blk, 0, stream>>>(
            TMPA, w16[12], P(13), TMPB, hdr, b, 0, 0, 0, 64, 1, 3, 1, 0, 0, -1);
        k_sigmoid_out<<<cdiv(hw0, 256), 256, 0, stream>>>(TMPB, hdr, b, dout);
    }
}
